// AttentionPool_7215545057869
// MI455X (gfx1250) — matrix operand order repaired and checked by reading, not compiled, not run on hardware
//
#include <hip/hip_runtime.h>
#include <hip/hip_bf16.h>

#define DIM 1024
#define HEADS 16
#define DH 64
#define NQ 128
#define NK 4096
#define NB 8

typedef __attribute__((ext_vector_type(8)))  __bf16 v8bf;
typedef __attribute__((ext_vector_type(16))) __bf16 v16bf;
typedef __attribute__((ext_vector_type(8)))  float  v8f;
typedef __attribute__((ext_vector_type(4)))  unsigned int u32x4;
typedef __attribute__((ext_vector_type(8)))  int i32x8;
typedef __attribute__((ext_vector_type(4)))  int i32x4;

#if defined(__AMDGCN__) && __has_builtin(__builtin_amdgcn_tensor_load_to_lds) && __has_builtin(__builtin_amdgcn_s_wait_tensorcnt)
#define USE_TDM 1
#else
#define USE_TDM 0
#endif

// A-fragment for 16x32 bf16 WMMA: lane L holds row M=L%16; K-base = (L/16)*16,
// elements 0..15 at kb..kb+15 (two 16B chunks).
__device__ __forceinline__ v16bf afrag(const __bf16* p) {
  v8bf lo = *(const v8bf*)p;
  v8bf hi = *(const v8bf*)(p + 8);
  return __builtin_shufflevector(lo, hi, 0,1,2,3,4,5,6,7,8,9,10,11,12,13,14,15);
}

__device__ __forceinline__ v8f wmma_bf16(v16bf a, v16bf b, v8f c) {
  return __builtin_amdgcn_wmma_f32_16x16x32_bf16(false, a, false, b, (short)0, c, false, false);
}

#if USE_TDM
// 1-D contiguous TDM load: `elems` bf16 elements from global -> LDS.
// D# per CDNA5 ISA 8.3/8.4: group0 = {flags/count, lds_addr, global_addr, type=2},
// group1 packs data_size=2B, tensor_dim0=elems, tensor_dim1=1, tile_dim0=elems,
// tile_dim1=1, tensor_dim0_stride=elems. Groups 2/3 zero (unused higher dims).
// This toolchain uses the 6-arg builtin form (extra int32x8 group, then cpol).
__device__ __forceinline__ void tdm_load_1d(const void* gptr, void* lptr, unsigned elems) {
  unsigned long long ga = (unsigned long long)(size_t)gptr;
  unsigned int la = (unsigned int)(size_t)lptr;   // generic addr low 32 bits = LDS offset
  u32x4 g0 = { 1u, la, (unsigned int)ga, (unsigned int)(ga >> 32) | (2u << 30) };
  i32x8 g1 = { (int)0x00010000,        // data_size=1 (2 bytes)
               (int)(elems << 16),     // tensor_dim0[15:0] in bits[63:48]
               (int)(1u << 16),        // tensor_dim0 hi=0, tensor_dim1 lo=1
               (int)(elems << 16),     // tensor_dim1 hi=0, tile_dim0 in bits[127:112]
               (int)1,                 // tile_dim1=1, tile_dim2=0
               (int)elems,             // tensor_dim0_stride low 32
               0, 0 };
  i32x4 z4 = { 0, 0, 0, 0 };
  i32x8 z8 = { 0, 0, 0, 0, 0, 0, 0, 0 };
  __builtin_amdgcn_tensor_load_to_lds(g0, g1, z4, z4, z8, 0);
}
#endif

// ---------------------------------------------------------------- converts
__global__ void cvt_f32_bf16(const float* __restrict__ s, __bf16* __restrict__ d, int n) {
  int i = blockIdx.x * blockDim.x + threadIdx.x;
  int stride = gridDim.x * blockDim.x;
  for (; i < n; i += stride) d[i] = (__bf16)s[i];
}

// ---------------------------------------------------------------- layernorm
__global__ void ln_kernel(const float* __restrict__ q, const float* __restrict__ w,
                          __bf16* __restrict__ out) {
  const int row = blockIdx.x;
  const int t = threadIdx.x;
  const float* src = q + (size_t)row * DIM;
  float x[4];
  float s1 = 0.f, s2 = 0.f;
#pragma unroll
  for (int i = 0; i < 4; i++) {
    x[i] = src[t + i * 256];
    s1 += x[i];
    s2 += x[i] * x[i];
  }
#pragma unroll
  for (int off = 16; off; off >>= 1) {
    s1 += __shfl_xor(s1, off, 32);
    s2 += __shfl_xor(s2, off, 32);
  }
  __shared__ float red1[8], red2[8];
  const int wv = t >> 5, lane = t & 31;
  if (lane == 0) { red1[wv] = s1; red2[wv] = s2; }
  __syncthreads();
  s1 = 0.f; s2 = 0.f;
#pragma unroll
  for (int i = 0; i < 8; i++) { s1 += red1[i]; s2 += red2[i]; }
  const float mean = s1 * (1.f / DIM);
  const float var  = s2 * (1.f / DIM) - mean * mean;
  const float inv  = rsqrtf(var + 1e-5f);
#pragma unroll
  for (int i = 0; i < 4; i++) {
    const int c = t + i * 256;
    out[(size_t)row * DIM + c] = (__bf16)((x[i] - mean) * inv * w[c]);
  }
}

// Shared B-tile staging: 32(k) x 64(n) source rows -> transposed LDS [n][k].
// One global_load_b128 (8 contiguous n) + 8 scattered ds_store_b16 per thread.
__device__ __forceinline__ void stage_B(__bf16* sB, const __bf16* __restrict__ Bw,
                                        int ldb, int kk, int n0, int tid) {
  const int r  = tid >> 3;          // k row 0..31
  const int c0 = (tid & 7) * 8;     // n col base
  union { uint4 u; __bf16 hv[8]; } tmp;
  tmp.u = *(const uint4*)&Bw[(size_t)(kk + r) * ldb + n0 + c0];
#pragma unroll
  for (int j = 0; j < 8; j++) sB[(c0 + j) * 32 + r] = tmp.hv[j];
}

// ---------------------------------------------------------------- GEMM: Q = qn @ Wq, + head RMSNorm
__global__ void gemm_q_kernel(const __bf16* __restrict__ A, const __bf16* __restrict__ Bw,
                              const float* __restrict__ gamma, __bf16* __restrict__ Qh) {
  __shared__ alignas(32) __bf16 sA[64 * 32];
  __shared__ alignas(32) __bf16 sB[64 * 32];
  __shared__ float sC[64 * 64];
  const int tid = threadIdx.x, lane = tid & 31, wv = tid >> 5;
  const int wm = wv & 3, wn = wv >> 2;
  const int m0 = blockIdx.x * 64;
  const int head = blockIdx.y;
  const int n0 = head * 64;
  v8f acc0 = {}, acc1 = {};
  for (int kk = 0; kk < DIM; kk += 32) {
    __syncthreads();
    {  // vectorized A staging: one 16B chunk per thread (64 rows x 4 chunks)
      int r = tid >> 2, c0 = (tid & 3) * 8;
      *(uint4*)&sA[r * 32 + c0] = *(const uint4*)&A[(size_t)(m0 + r) * DIM + kk + c0];
    }
    stage_B(sB, Bw, DIM, kk, n0, tid);
    __syncthreads();
    const int arow = wm * 16 + (lane & 15);
    const int akb  = (lane >> 4) * 16;
    v16bf a = afrag(&sA[arow * 32 + akb]);
    const int bkb = (lane >> 4) * 16;
    {
      int n = wn * 32 + (lane & 15);
      v16bf b = *(const v16bf*)&sB[n * 32 + bkb];
      acc0 = wmma_bf16(a, b, acc0);
    }
    {
      int n = wn * 32 + 16 + (lane & 15);
      v16bf b = *(const v16bf*)&sB[n * 32 + bkb];
      acc1 = wmma_bf16(a, b, acc1);
    }
  }
  __syncthreads();
  {
    const int row = wm * 16 + (lane >> 4) * 8;
    const int col = wn * 32 + (lane & 15);
#pragma unroll
    for (int r = 0; r < 8; r++) {
      sC[(row + r) * 64 + col]      = acc0[r];
      sC[(row + r) * 64 + col + 16] = acc1[r];
    }
  }
  __syncthreads();
  {  // rmsnorm epilogue: 4 lanes per row
    const int row = tid >> 2;
    const int qd = (tid & 3) * 16;
    float ss = 0.f;
#pragma unroll
    for (int j = 0; j < 16; j++) { float v = sC[row * 64 + qd + j]; ss += v * v; }
    ss += __shfl_xor(ss, 1, 4);
    ss += __shfl_xor(ss, 2, 4);
    const float inv = 8.0f / fmaxf(sqrtf(ss), 1e-12f);
    const int grow = m0 + row;
    const int bb = grow >> 7, nn = grow & 127;
    __bf16* dst = Qh + ((size_t)(bb * HEADS + head) * NQ + nn) * DH;
#pragma unroll
    for (int j = 0; j < 16; j++) {
      int d = qd + j;
      dst[d] = (__bf16)(sC[row * 64 + d] * inv * gamma[head * DH + d]);
    }
  }
}

// ---------------------------------------------------------------- GEMM: KVp = kv @ Wkv (A converted in staging)
__global__ void gemm_kv_kernel(const float* __restrict__ A, const __bf16* __restrict__ Bw,
                               const float* __restrict__ gammaK,
                               __bf16* __restrict__ Kh, __bf16* __restrict__ Vh) {
  __shared__ alignas(32) __bf16 sA[64 * 32];
  __shared__ alignas(32) __bf16 sB[64 * 32];
  __shared__ float sC[64 * 64];
  const int tid = threadIdx.x, lane = tid & 31, wv = tid >> 5;
  const int wm = wv & 3, wn = wv >> 2;
  const int m0 = blockIdx.x * 64;
  const int cblk = blockIdx.y;
  const int n0 = cblk * 64;
  v8f acc0 = {}, acc1 = {};
  for (int kk = 0; kk < DIM; kk += 32) {
    __syncthreads();
    {  // A staging with f32->bf16 packing: 8 values per thread
      int r = tid >> 2, c0 = (tid & 3) * 8;
      const float* src = &A[(size_t)(m0 + r) * DIM + kk + c0];
      v8bf v;
#pragma unroll
      for (int j = 0; j < 8; j++) v[j] = (__bf16)src[j];
      *(v8bf*)&sA[r * 32 + c0] = v;
      if (kk + 32 < DIM)  // keep the L2-resident kv stream ahead of the WMMA pipe
        __builtin_prefetch(src + 32, 0, 3);
    }
    stage_B(sB, Bw, 2 * DIM, kk, n0, tid);
    __syncthreads();
    const int arow = wm * 16 + (lane & 15);
    const int akb  = (lane >> 4) * 16;
    v16bf a = afrag(&sA[arow * 32 + akb]);
    const int bkb = (lane >> 4) * 16;
    {
      int n = wn * 32 + (lane & 15);
      v16bf b = *(const v16bf*)&sB[n * 32 + bkb];
      acc0 = wmma_bf16(a, b, acc0);
    }
    {
      int n = wn * 32 + 16 + (lane & 15);
      v16bf b = *(const v16bf*)&sB[n * 32 + bkb];
      acc1 = wmma_bf16(a, b, acc1);
    }
  }
  __syncthreads();
  {
    const int row = wm * 16 + (lane >> 4) * 8;
    const int col = wn * 32 + (lane & 15);
#pragma unroll
    for (int r = 0; r < 8; r++) {
      sC[(row + r) * 64 + col]      = acc0[r];
      sC[(row + r) * 64 + col + 16] = acc1[r];
    }
  }
  __syncthreads();
  {
    const int row = tid >> 2;
    const int qd = (tid & 3) * 16;
    const int grow = m0 + row;
    const int bb = grow >> 12, mm = grow & 4095;
    if (cblk < HEADS) {  // K half -> rmsnorm
      float ss = 0.f;
#pragma unroll
      for (int j = 0; j < 16; j++) { float v = sC[row * 64 + qd + j]; ss += v * v; }
      ss += __shfl_xor(ss, 1, 4);
      ss += __shfl_xor(ss, 2, 4);
      const float inv = 8.0f / fmaxf(sqrtf(ss), 1e-12f);
      const int head = cblk;
      __bf16* dst = Kh + ((size_t)(bb * HEADS + head) * NK + mm) * DH;
#pragma unroll
      for (int j = 0; j < 16; j++) {
        int d = qd + j;
        dst[d] = (__bf16)(sC[row * 64 + d] * inv * gammaK[head * DH + d]);
      }
    } else {             // V half -> plain store
      const int head = cblk - HEADS;
      __bf16* dst = Vh + ((size_t)(bb * HEADS + head) * NK + mm) * DH;
#pragma unroll
      for (int j = 0; j < 16; j++) {
        int d = qd + j;
        dst[d] = (__bf16)sC[row * 64 + d];
      }
    }
  }
}

// ---------------------------------------------------------------- flash attention, one block per (b,h)
__global__ void attn_kernel(const __bf16* __restrict__ Qh, const __bf16* __restrict__ Kh,
                            const __bf16* __restrict__ Vh, const unsigned char* __restrict__ mask,
                            __bf16* __restrict__ O) {
  __shared__ alignas(32) __bf16 sQ[NQ * DH];
  __shared__ alignas(32) __bf16 sK[64 * DH];
  __shared__ alignas(32) __bf16 sVt[DH * 64];
  __shared__ alignas(32) __bf16 sP[8 * 16 * 64];
  __shared__ float sBias[64];
  const int tid = threadIdx.x, lane = tid & 31, wv = tid >> 5;
  const int bh = blockIdx.x, b = bh >> 4, h = bh & 15;
  // stage all of Q for this (b,h): 128x64 bf16 = 16KB contiguous
#if USE_TDM
  if (wv == 0) {
    tdm_load_1d(Qh + (size_t)bh * NQ * DH, sQ, NQ * DH);
    __builtin_amdgcn_s_wait_tensorcnt(0);
  }
#else
  {
    const uint4* src = (const uint4*)(Qh + (size_t)bh * NQ * DH);
    uint4* dst = (uint4*)sQ;
#pragma unroll
    for (int i = 0; i < 4; i++) dst[i * 256 + tid] = src[i * 256 + tid];
  }
#endif
  __syncthreads();
  const int rbase = (lane >> 4) * 8;
  v16bf qA0, qA1;
  {
    const int row = wv * 16 + (lane & 15);
    qA0 = afrag(&sQ[row * DH + (lane >> 4) * 16]);
    qA1 = afrag(&sQ[row * DH + 32 + (lane >> 4) * 16]);
  }
  v8f o0 = {}, o1 = {}, o2 = {}, o3 = {};
  float rmax[8], rsum[8];
#pragma unroll
  for (int r = 0; r < 8; r++) { rmax[r] = -3.0e38f; rsum[r] = 0.f; }
  const __bf16* Kbase = Kh + (size_t)bh * NK * DH;
  const __bf16* Vbase = Vh + (size_t)bh * NK * DH;
  const int bkb = (lane >> 4) * 16;

  for (int t0 = 0; t0 < NK; t0 += 64) {
    __syncthreads();
    // stage K tile [key][d]: 64x64 bf16 = 8KB contiguous
#if USE_TDM
    if (wv == 0) tdm_load_1d(Kbase + (size_t)t0 * DH, sK, 64 * DH);  // overlaps V staging below
#else
    {
      const uint4* src = (const uint4*)(Kbase + (size_t)t0 * DH);
      uint4* dst = (uint4*)sK;
      dst[tid] = src[tid];
      dst[256 + tid] = src[256 + tid];
    }
#endif
    {  // stage V tile transposed: sVt[d][key]
      const uint4* src = (const uint4*)(Vbase + (size_t)t0 * DH);
#pragma unroll
      for (int i = 0; i < 2; i++) {
        int c = i * 256 + tid;
        union { uint4 u; __bf16 hv[8]; } tmp;
        tmp.u = src[c];
        int key = c >> 3, d0 = (c & 7) * 8;
#pragma unroll
        for (int j = 0; j < 8; j++) sVt[(d0 + j) * 64 + key] = tmp.hv[j];
      }
    }
    if (tid < 64) sBias[tid] = mask[b * NK + t0 + tid] ? 0.0f : -3.0e38f;
    if (t0 + 64 < NK) __builtin_prefetch(Vbase + (size_t)(t0 + 64) * DH + tid * 8, 0, 3);
#if USE_TDM
    if (wv == 0) __builtin_amdgcn_s_wait_tensorcnt(0);
#endif
    __syncthreads();

    // ---- S = Q K^T  (4 key sub-tiles x 2 k-steps over d)
    v8f S0 = {}, S1 = {}, S2 = {}, S3 = {};
    {
      int key = (lane & 15);
      v16bf b0 = *(const v16bf*)&sK[key * DH + bkb];
      v16bf b1 = *(const v16bf*)&sK[key * DH + 32 + bkb];
      S0 = wmma_bf16(qA0, b0, S0); S0 = wmma_bf16(qA1, b1, S0);
      key = 16 + (lane & 15);
      b0 = *(const v16bf*)&sK[key * DH + bkb];
      b1 = *(const v16bf*)&sK[key * DH + 32 + bkb];
      S1 = wmma_bf16(qA0, b0, S1); S1 = wmma_bf16(qA1, b1, S1);
      key = 32 + (lane & 15);
      b0 = *(const v16bf*)&sK[key * DH + bkb];
      b1 = *(const v16bf*)&sK[key * DH + 32 + bkb];
      S2 = wmma_bf16(qA0, b0, S2); S2 = wmma_bf16(qA1, b1, S2);
      key = 48 + (lane & 15);
      b0 = *(const v16bf*)&sK[key * DH + bkb];
      b1 = *(const v16bf*)&sK[key * DH + 32 + bkb];
      S3 = wmma_bf16(qA0, b0, S3); S3 = wmma_bf16(qA1, b1, S3);
    }
    // ---- online softmax (row stats replicated across the 16 lanes of a half-wave)
    const float bias0 = sBias[(lane & 15)];
    const float bias1 = sBias[16 + (lane & 15)];
    const float bias2 = sBias[32 + (lane & 15)];
    const float bias3 = sBias[48 + (lane & 15)];
#pragma unroll
    for (int r = 0; r < 8; r++) {
      float v0 = S0[r] + bias0, v1 = S1[r] + bias1, v2 = S2[r] + bias2, v3 = S3[r] + bias3;
      float mx = fmaxf(fmaxf(v0, v1), fmaxf(v2, v3));
      mx = fmaxf(mx, __shfl_xor(mx, 1, 16));
      mx = fmaxf(mx, __shfl_xor(mx, 2, 16));
      mx = fmaxf(mx, __shfl_xor(mx, 4, 16));
      mx = fmaxf(mx, __shfl_xor(mx, 8, 16));
      const float nm = fmaxf(rmax[r], mx);
      const float corr = __expf(rmax[r] - nm);
      rmax[r] = nm;
      v0 = __expf(v0 - nm); v1 = __expf(v1 - nm); v2 = __expf(v2 - nm); v3 = __expf(v3 - nm);
      float ps = v0 + v1 + v2 + v3;
      ps += __shfl_xor(ps, 1, 16);
      ps += __shfl_xor(ps, 2, 16);
      ps += __shfl_xor(ps, 4, 16);
      ps += __shfl_xor(ps, 8, 16);
      rsum[r] = rsum[r] * corr + ps;
      o0[r] *= corr; o1[r] *= corr; o2[r] *= corr; o3[r] *= corr;
      // scatter P row to per-wave LDS scratch (C-layout -> row-major)
      __bf16* pp = &sP[(wv * 16 + rbase + r) * 64 + (lane & 15)];
      pp[0]  = (__bf16)v0;
      pp[16] = (__bf16)v1;
      pp[32] = (__bf16)v2;
      pp[48] = (__bf16)v3;
    }
    // re-read P as A-fragments (same-wave LDS region, DS ops in order)
    v16bf pA0, pA1;
    {
      const int row = wv * 16 + (lane & 15);
      pA0 = afrag(&sP[row * 64 + (lane >> 4) * 16]);
      pA1 = afrag(&sP[row * 64 + 32 + (lane >> 4) * 16]);
    }
    // ---- O += P V  (4 d sub-tiles x 2 k-steps over keys)
    {
      int d = (lane & 15);
      v16bf b0 = *(const v16bf*)&sVt[d * 64 + bkb];
      v16bf b1 = *(const v16bf*)&sVt[d * 64 + 32 + bkb];
      o0 = wmma_bf16(pA0, b0, o0); o0 = wmma_bf16(pA1, b1, o0);
      d = 16 + (lane & 15);
      b0 = *(const v16bf*)&sVt[d * 64 + bkb];
      b1 = *(const v16bf*)&sVt[d * 64 + 32 + bkb];
      o1 = wmma_bf16(pA0, b0, o1); o1 = wmma_bf16(pA1, b1, o1);
      d = 32 + (lane & 15);
      b0 = *(const v16bf*)&sVt[d * 64 + bkb];
      b1 = *(const v16bf*)&sVt[d * 64 + 32 + bkb];
      o2 = wmma_bf16(pA0, b0, o2); o2 = wmma_bf16(pA1, b1, o2);
      d = 48 + (lane & 15);
      b0 = *(const v16bf*)&sVt[d * 64 + bkb];
      b1 = *(const v16bf*)&sVt[d * 64 + 32 + bkb];
      o3 = wmma_bf16(pA0, b0, o3); o3 = wmma_bf16(pA1, b1, o3);
    }
  }
  // ---- epilogue: normalize and store merged-head layout (b, n, h*64+d) bf16
  {
    const int nrow = wv * 16 + rbase;
#pragma unroll
    for (int r = 0; r < 8; r++) {
      const float inv = 1.0f / rsum[r];
      const int n = nrow + r;
      __bf16* dst = O + ((size_t)(b * NQ + n)) * DIM + h * DH + (lane & 15);
      dst[0]  = (__bf16)(o0[r] * inv);
      dst[16] = (__bf16)(o1[r] * inv);
      dst[32] = (__bf16)(o2[r] * inv);
      dst[48] = (__bf16)(o3[r] * inv);
    }
  }
}

// ---------------------------------------------------------------- GEMM: final = attn @ Wout (f32 store)
__global__ void gemm_out_kernel(const __bf16* __restrict__ A, const __bf16* __restrict__ Bw,
                                float* __restrict__ out) {
  __shared__ alignas(32) __bf16 sA[64 * 32];
  __shared__ alignas(32) __bf16 sB[64 * 32];
  const int tid = threadIdx.x, lane = tid & 31, wv = tid >> 5;
  const int wm = wv & 3, wn = wv >> 2;
  const int m0 = blockIdx.x * 64;
  const int n0 = blockIdx.y * 64;
  v8f acc0 = {}, acc1 = {};
  for (int kk = 0; kk < DIM; kk += 32) {
    __syncthreads();
    {
      int r = tid >> 2, c0 = (tid & 3) * 8;
      *(uint4*)&sA[r * 32 + c0] = *(const uint4*)&A[(size_t)(m0 + r) * DIM + kk + c0];
    }
    stage_B(sB, Bw, DIM, kk, n0, tid);
    __syncthreads();
    const int arow = wm * 16 + (lane & 15);
    const int akb  = (lane >> 4) * 16;
    v16bf a = afrag(&sA[arow * 32 + akb]);
    const int bkb = (lane >> 4) * 16;
    {
      int n = wn * 32 + (lane & 15);
      v16bf b = *(const v16bf*)&sB[n * 32 + bkb];
      acc0 = wmma_bf16(a, b, acc0);
    }
    {
      int n = wn * 32 + 16 + (lane & 15);
      v16bf b = *(const v16bf*)&sB[n * 32 + bkb];
      acc1 = wmma_bf16(a, b, acc1);
    }
  }
  const int row0 = m0 + wm * 16 + (lane >> 4) * 8;
  const int col  = n0 + wn * 32 + (lane & 15);
#pragma unroll
  for (int r = 0; r < 8; r++) {
    out[(size_t)(row0 + r) * DIM + col]      = acc0[r];
    out[(size_t)(row0 + r) * DIM + col + 16] = acc1[r];
  }
}

// ----------------------------------------------------------------
extern "C" void kernel_launch(void* const* d_in, const int* in_sizes, int n_in,
                              void* d_out, int out_size, void* d_ws, size_t ws_size,
                              hipStream_t stream) {
  (void)in_sizes; (void)n_in; (void)out_size; (void)ws_size;
  const float* q    = (const float*)d_in[0];
  const float* kv   = (const float*)d_in[1];
  const unsigned char* mask = (const unsigned char*)d_in[2];
  const float* ln_w = (const float*)d_in[3];
  const float* gq   = (const float*)d_in[4];
  const float* gk   = (const float*)d_in[5];
  const float* Wq   = (const float*)d_in[6];
  const float* Wkv  = (const float*)d_in[7];
  const float* Wout = (const float*)d_in[8];
  float* out = (float*)d_out;

  size_t off = 0;
  char* base = (char*)d_ws;
  auto alloc = [&](size_t bytes) -> void* {
    void* p = base + off;
    off += (bytes + 255) & ~(size_t)255;
    return p;
  };
  __bf16* qn_bf   = (__bf16*)alloc((size_t)1024 * 1024 * 2);           // LN(q), bf16
  __bf16* Wq_bf   = (__bf16*)alloc((size_t)1024 * 1024 * 2);
  __bf16* Wkv_bf  = (__bf16*)alloc((size_t)1024 * 2048 * 2);
  __bf16* Wout_bf = (__bf16*)alloc((size_t)1024 * 1024 * 2);
  __bf16* Qh      = (__bf16*)alloc((size_t)NB * HEADS * NQ * DH * 2);  // (b,h,n,d)
  __bf16* Kh      = (__bf16*)alloc((size_t)NB * HEADS * NK * DH * 2);  // (b,h,m,d)
  __bf16* Vh      = (__bf16*)alloc((size_t)NB * HEADS * NK * DH * 2);  // (b,h,m,d)
  __bf16* Obf     = (__bf16*)alloc((size_t)1024 * 1024 * 2);           // merged heads (b,n,inner)

  cvt_f32_bf16<<<1024, 256, 0, stream>>>(Wq,   Wq_bf,   1024 * 1024);
  cvt_f32_bf16<<<2048, 256, 0, stream>>>(Wkv,  Wkv_bf,  1024 * 2048);
  cvt_f32_bf16<<<1024, 256, 0, stream>>>(Wout, Wout_bf, 1024 * 1024);
  ln_kernel<<<1024, 256, 0, stream>>>(q, ln_w, qn_bf);
  gemm_q_kernel<<<dim3(16, 16), 256, 0, stream>>>(qn_bf, Wq_bf, gq, Qh);
  gemm_kv_kernel<<<dim3(512, 32), 256, 0, stream>>>(kv, Wkv_bf, gk, Kh, Vh);
  attn_kernel<<<128, 256, 0, stream>>>(Qh, Kh, Vh, mask, Obf);
  gemm_out_kernel<<<dim3(16, 16), 256, 0, stream>>>(Obf, Wout_bf, out);
}
